// MambaBlock_20383914787431
// MI455X (gfx1250) — matrix operand order repaired and checked by reading, not compiled, not run on hardware
//
#include <hip/hip_runtime.h>
#include <cstddef>

// ---- problem constants (match reference) ----
#define B_ 2
#define L_ 1024
#define D_ 1024
#define ED_ 2048        // EXPAND * D
#define N_ 16
#define R_ 64
#define KC_ 4           // conv kernel
#define BL_ (B_ * L_)   // 2048 rows
#define DBC_PAD 128     // R + 2N = 96 padded to 128 (multiple of 64)

typedef __attribute__((ext_vector_type(16))) __bf16 v16bf;
typedef __attribute__((ext_vector_type(8)))  __bf16 v8bf;
typedef __attribute__((ext_vector_type(8)))  float  v8f;

__device__ __forceinline__ unsigned short f2bf(float f) {
  unsigned int u = __float_as_uint(f);
  unsigned int r = u + 0x7FFFu + ((u >> 16) & 1u);   // round-to-nearest-even
  return (unsigned short)(r >> 16);
}

// A fragment (16x32 bf16): lane(half,m) holds K=[half*8..+7] and [16+half*8..+7]
__device__ __forceinline__ v16bf load_a_frag(const __bf16* row, int k0, int half) {
  v8bf a0 = *(const v8bf*)(row + k0 + half * 8);
  v8bf a1 = *(const v8bf*)(row + k0 + 16 + half * 8);
  v16bf a;
#pragma unroll
  for (int i = 0; i < 8; ++i) { a[i] = a0[i]; a[8 + i] = a1[i]; }
  return a;
}

// ---------------- f32 -> bf16 conversion ----------------
__global__ void cvt_f32_bf16(const float* __restrict__ in,
                             unsigned short* __restrict__ out, int n) {
  int i = blockIdx.x * blockDim.x + threadIdx.x;
  if (i < n) out[i] = f2bf(in[i]);
}

// f32 (rowsIn x cols) -> bf16 (rowsOut x cols), zero-pad extra rows
__global__ void cvt_pad_bf16(const float* __restrict__ in,
                             unsigned short* __restrict__ out,
                             int rowsIn, int rowsOut, int cols) {
  int i = blockIdx.x * blockDim.x + threadIdx.x;
  if (i < rowsOut * cols) {
    int r = i / cols;
    out[i] = (r < rowsIn) ? f2bf(in[i]) : (unsigned short)0;
  }
}

// extract dt slice dbc[:, 0:R] -> bf16 (BL x R); dbc has row stride DBC_PAD
__global__ void extract_dt_bf16(const float* __restrict__ dbc,
                                unsigned short* __restrict__ out) {
  int i = blockIdx.x * blockDim.x + threadIdx.x;
  if (i < BL_ * R_) {
    int row = i >> 6, c = i & 63;
    out[i] = f2bf(dbc[row * DBC_PAD + c]);
  }
}

// ------- bf16 WMMA GEMM:  C(M,N) = A(M,K) * W(N,K)^T (+bias)(+act) ---------
// One wave computes a 32(M) x 64(N) tile: 2 A-frags x 4 B-frags = 8 WMMAs/k.
// Requires: M % 32 == 0, Nout % 64 == 0, Kd % 32 == 0.   act: 0=none 1=softplus
__global__ void gemm_bf16_wmma(const unsigned short* __restrict__ A_,
                               const unsigned short* __restrict__ W_,
                               const float* __restrict__ bias,
                               float* __restrict__ C,
                               int M, int Kd, int Nout, int act) {
  const __bf16* A = reinterpret_cast<const __bf16*>(A_);
  const __bf16* W = reinterpret_cast<const __bf16*>(W_);

  int wave = blockIdx.x * (blockDim.x >> 5) + (threadIdx.x >> 5);
  int lane = threadIdx.x & 31;
  int ntiles = Nout >> 6;
  int mtiles = M >> 5;
  if (wave >= mtiles * ntiles) return;   // exact launches: never divergent

  int mt = wave / ntiles;
  int nt = wave - mt * ntiles;
  int half = lane >> 4;                  // 0: lanes 0-15, 1: lanes 16-31
  int lm   = lane & 15;

  const __bf16* rowA0 = A + (size_t)(mt * 32 + lm) * Kd;
  const __bf16* rowA1 = rowA0 + (size_t)16 * Kd;
  const __bf16* rowW[4];
#pragma unroll
  for (int t = 0; t < 4; ++t)
    rowW[t] = W + (size_t)((nt << 6) + (t << 4) + lm) * Kd;

  v8f acc[8] = {};

  for (int k0 = 0; k0 < Kd; k0 += 32) {
    if (k0 + 128 < Kd) {
      __builtin_prefetch(rowA0 + k0 + 128, 0, 3);
      __builtin_prefetch(rowA1 + k0 + 128, 0, 3);
    }
    // batch all loads for this k-step, then issue the 8 WMMAs
    v16bf a0 = load_a_frag(rowA0, k0, half);
    v16bf a1 = load_a_frag(rowA1, k0, half);
    v16bf b[4];
#pragma unroll
    for (int t = 0; t < 4; ++t)
      b[t] = load_a_frag(rowW[t], k0, half);   // B: K=[half*8..+7],[16+half*8..+7], col=lm

#pragma unroll
    for (int t = 0; t < 4; ++t)
      acc[t] = __builtin_amdgcn_wmma_f32_16x16x32_bf16(
          false, a0, false, b[t], (short)0, acc[t], false, false);
#pragma unroll
    for (int t = 0; t < 4; ++t)
      acc[4 + t] = __builtin_amdgcn_wmma_f32_16x16x32_bf16(
          false, a1, false, b[t], (short)0, acc[4 + t], false, false);
  }

  // epilogue: acc[t][r] -> C[mt*32 + r + 8*half][nt*64 + t*16 + lm] (+16 rows for acc[4+t])
#pragma unroll
  for (int t = 0; t < 4; ++t) {
    int col = (nt << 6) + (t << 4) + lm;
    float bv = bias ? bias[col] : 0.0f;
#pragma unroll
    for (int r = 0; r < 8; ++r) {
      int m0 = (mt << 5) + r + (half << 3);
      float v0 = acc[t][r] + bv;
      float v1 = acc[4 + t][r] + bv;
      if (act == 1) {
        v0 = (v0 > 20.0f) ? v0 : log1pf(__expf(v0));
        v1 = (v1 > 20.0f) ? v1 : log1pf(__expf(v1));
      }
      C[(size_t)m0 * Nout + col] = v0;
      C[(size_t)(m0 + 16) * Nout + col] = v1;
    }
  }
}

// ---------------- causal depthwise conv (K=4) + bias + SiLU ----------------
__global__ void conv_silu(const float* __restrict__ xz,
                          const float* __restrict__ conv_w,
                          const float* __restrict__ conv_b,
                          float* __restrict__ xc,
                          unsigned short* __restrict__ xcbf) {
  int idx = blockIdx.x * blockDim.x + threadIdx.x;
  if (idx >= BL_ * ED_) return;
  int e   = idx % ED_;
  int row = idx / ED_;          // b*L + l
  int l   = row % L_;
  int b   = row / L_;
  float s = conv_b[e];
#pragma unroll
  for (int k = 0; k < KC_; ++k) {
    int ls = l + k - (KC_ - 1);
    if (ls >= 0)
      s += xz[(size_t)(b * L_ + ls) * (2 * ED_) + e] * conv_w[e * KC_ + k];
  }
  float v = s / (1.0f + __expf(-s));   // SiLU
  xc[idx] = v;
  xcbf[idx] = f2bf(v);
}

// ---------------- selective scan + gating ----------------
__global__ void selective_scan(const float* __restrict__ dtf,   // (BL, ED)
                               const float* __restrict__ xc,    // (BL, ED)
                               const float* __restrict__ xz,    // (BL, 2ED) - z half
                               const float* __restrict__ dbc,   // (BL, DBC_PAD)
                               const float* __restrict__ A_log, // (ED, N)
                               const float* __restrict__ Dp,    // (ED)
                               unsigned short* __restrict__ yg) // (BL, ED) bf16
{
  int idx = blockIdx.x * blockDim.x + threadIdx.x;
  if (idx >= B_ * ED_) return;
  int b = idx / ED_;
  int e = idx % ED_;

  float Arow[N_], h[N_];
#pragma unroll
  for (int n = 0; n < N_; ++n) {
    Arow[n] = -__expf(A_log[e * N_ + n]);
    h[n] = 0.0f;
  }
  float Dpe = Dp[e];

  for (int l = 0; l < L_; ++l) {
    int row = b * L_ + l;
    float dtv = dtf[(size_t)row * ED_ + e];
    float xv  = xc [(size_t)row * ED_ + e];
    float zv  = xz [(size_t)row * (2 * ED_) + ED_ + e];
    float dx  = dtv * xv;
    const float* bc = dbc + (size_t)row * DBC_PAD;
    float acc = 0.0f;
#pragma unroll
    for (int n = 0; n < N_; ++n) {
      float dA = __expf(dtv * Arow[n]);
      h[n] = dA * h[n] + dx * bc[R_ + n];        // B_n
      acc += bc[R_ + N_ + n] * h[n];             // C_n
    }
    float y  = acc + Dpe * xv;
    float sz = zv / (1.0f + __expf(-zv));        // silu(z)
    yg[(size_t)row * ED_ + e] = f2bf(y * sz);
  }
}

// ---------------- launch ----------------
extern "C" void kernel_launch(void* const* d_in, const int* in_sizes, int n_in,
                              void* d_out, int out_size, void* d_ws, size_t ws_size,
                              hipStream_t stream) {
  const float* x         = (const float*)d_in[0];   // (B,L,D)
  const float* in_proj_w = (const float*)d_in[1];   // (2ED, D)
  const float* in_proj_b = (const float*)d_in[2];   // (2ED)
  const float* conv_w    = (const float*)d_in[3];   // (ED,1,K)
  const float* conv_b    = (const float*)d_in[4];   // (ED)
  const float* x_proj_w  = (const float*)d_in[5];   // (R+2N, ED)
  const float* dt_proj_w = (const float*)d_in[6];   // (ED, R)
  const float* dt_proj_b = (const float*)d_in[7];   // (ED)
  const float* A_log     = (const float*)d_in[8];   // (ED, N)
  const float* D_param   = (const float*)d_in[9];   // (ED)
  const float* out_proj_w= (const float*)d_in[10];  // (D, ED)
  float* out = (float*)d_out;                       // (B,L,D)

  // workspace carve-up (256B aligned)
  char* base = (char*)d_ws;
  size_t off = 0;
  auto carve = [&](size_t bytes) -> char* {
    char* p = base + off;
    off = (off + bytes + 255) & ~(size_t)255;
    return p;
  };
  unsigned short* xbf  = (unsigned short*)carve((size_t)BL_ * D_ * 2);
  unsigned short* w1bf = (unsigned short*)carve((size_t)2 * ED_ * D_ * 2);
  unsigned short* w2bf = (unsigned short*)carve((size_t)DBC_PAD * ED_ * 2);   // padded
  unsigned short* w3bf = (unsigned short*)carve((size_t)ED_ * R_ * 2);
  unsigned short* w4bf = (unsigned short*)carve((size_t)D_ * ED_ * 2);
  float* xz   = (float*)carve((size_t)BL_ * 2 * ED_ * 4);
  float* xc   = (float*)carve((size_t)BL_ * ED_ * 4);
  unsigned short* xcbf = (unsigned short*)carve((size_t)BL_ * ED_ * 2);
  float* dbc  = (float*)carve((size_t)BL_ * DBC_PAD * 4);                     // padded
  unsigned short* dtp  = (unsigned short*)carve((size_t)BL_ * R_ * 2);
  float* dtf  = (float*)carve((size_t)BL_ * ED_ * 4);
  unsigned short* ygbf = (unsigned short*)carve((size_t)BL_ * ED_ * 2);
  (void)ws_size;

  const int T = 256;
  auto cdiv = [](int a, int b) { return (a + b - 1) / b; };

  // 1) fp32 -> bf16 for GEMM operands
  cvt_f32_bf16<<<cdiv(BL_ * D_, T), T, 0, stream>>>(x, xbf, BL_ * D_);
  cvt_f32_bf16<<<cdiv(2 * ED_ * D_, T), T, 0, stream>>>(in_proj_w, w1bf, 2 * ED_ * D_);
  cvt_pad_bf16<<<cdiv(DBC_PAD * ED_, T), T, 0, stream>>>(x_proj_w, w2bf,
                                                         R_ + 2 * N_, DBC_PAD, ED_);
  cvt_f32_bf16<<<cdiv(ED_ * R_, T), T, 0, stream>>>(dt_proj_w, w3bf, ED_ * R_);
  cvt_f32_bf16<<<cdiv(D_ * ED_, T), T, 0, stream>>>(out_proj_w, w4bf, D_ * ED_);

  // 2) in_proj: xz = x @ in_proj_w^T + b      M=2048 K=1024 N=4096
  {
    int waves = (BL_ / 32) * ((2 * ED_) / 64);
    gemm_bf16_wmma<<<cdiv(waves, 8), T, 0, stream>>>(xbf, w1bf, in_proj_b, xz,
                                                     BL_, D_, 2 * ED_, 0);
  }

  // 3) causal depthwise conv + SiLU  -> xc (f32 + bf16)
  conv_silu<<<cdiv(BL_ * ED_, T), T, 0, stream>>>(xz, conv_w, conv_b, xc, xcbf);

  // 4) x_proj: dbc = xc @ x_proj_w^T           M=2048 K=2048 N=128 (padded)
  {
    int waves = (BL_ / 32) * (DBC_PAD / 64);
    gemm_bf16_wmma<<<cdiv(waves, 8), T, 0, stream>>>(xcbf, w2bf, nullptr, dbc,
                                                     BL_, ED_, DBC_PAD, 0);
  }

  // 5) dt slice -> bf16, then dt_proj + softplus  M=2048 K=64 N=2048
  extract_dt_bf16<<<cdiv(BL_ * R_, T), T, 0, stream>>>(dbc, dtp);
  {
    int waves = (BL_ / 32) * (ED_ / 64);
    gemm_bf16_wmma<<<cdiv(waves, 8), T, 0, stream>>>(dtp, w3bf, dt_proj_b, dtf,
                                                     BL_, R_, ED_, 1);
  }

  // 6) selective scan + D skip + silu(z) gating -> yg (bf16)
  selective_scan<<<cdiv(B_ * ED_, T), T, 0, stream>>>(dtf, xc, xz, dbc, A_log,
                                                      D_param, ygbf);

  // 7) out_proj: out = yg @ out_proj_w^T       M=2048 K=2048 N=1024
  {
    int waves = (BL_ / 32) * (D_ / 64);
    gemm_bf16_wmma<<<cdiv(waves, 8), T, 0, stream>>>(ygbf, w4bf, nullptr, out,
                                                     BL_, ED_, D_, 0);
  }
}
